// ConvLSTM_73495480369688
// MI455X (gfx1250) — matrix operand order repaired and checked by reading, not compiled, not run on hardware
//
#include <hip/hip_runtime.h>
#include <hip/hip_bf16.h>
#include <math.h>

typedef _Float16 f16;
typedef __attribute__((ext_vector_type(16))) _Float16 v16h;
typedef __attribute__((ext_vector_type(8)))  _Float16 v8h;
typedef __attribute__((ext_vector_type(8)))  float    v8f;

#define Bsz 128
#define Tsz 128
#define KV  1728      // C*27
#define NG  256       // 4*HID
#define MT  (Bsz*Tsz) // 16384

__device__ __forceinline__ int imax(int a, int b){ return a > b ? a : b; }
__device__ __forceinline__ int imin(int a, int b){ return a < b ? a : b; }
__device__ __forceinline__ float sigf(float x){ return 1.f/(1.f+__expf(-x)); }

// A-fragment (16x32 f16): elems 0..7 = p[0..7], 8..15 = p[16..23]
__device__ __forceinline__ v16h frag_a16(const f16* p){
  v8h lo = *(const v8h*)p;
  v8h hi = *(const v8h*)(p + 16);
  return __builtin_shufflevector(lo, hi, 0,1,2,3,4,5,6,7,8,9,10,11,12,13,14,15);
}
// B-fragment (32x16 f16) from K-contiguous storage: 16 contiguous halves
__device__ __forceinline__ v16h frag_b16(const f16* p){
  v8h lo = *(const v8h*)p;
  v8h hi = *(const v8h*)(p + 8);
  return __builtin_shufflevector(lo, hi, 0,1,2,3,4,5,6,7,8,9,10,11,12,13,14,15);
}

// ------------------- channel attention: att[row][c] -------------------
__global__ __launch_bounds__(64) void attn_kernel(const float* __restrict__ x,
    const float* __restrict__ fc1w, const float* __restrict__ fc1b,
    const float* __restrict__ fc2w, const float* __restrict__ fc2b,
    float* __restrict__ att)
{
  __shared__ float xa[64];
  __shared__ float a1[64];
  __shared__ float red[64];
  const int row = blockIdx.x;
  const int j = threadIdx.x;
  const float* p = x + (size_t)row*KV + j*27;
  float s = 0.f;
  #pragma unroll
  for (int k = 0; k < 27; ++k) s += p[k];
  xa[j] = s * (1.f/27.f);
  __syncthreads();
  float acc = fc1b[j];
  #pragma unroll 8
  for (int c = 0; c < 64; ++c) acc += xa[c] * fc1w[j*64 + c];
  a1[j] = tanhf(acc);
  __syncthreads();
  acc = fc2b[j];
  #pragma unroll 8
  for (int c = 0; c < 64; ++c) acc += a1[c] * fc2w[j*64 + c];
  __syncthreads();
  xa[j] = acc;
  red[j] = acc;
  __syncthreads();
  for (int off = 32; off > 0; off >>= 1){
    if (j < off) red[j] = fmaxf(red[j], red[j+off]);
    __syncthreads();
  }
  const float m = red[0];
  __syncthreads();
  const float e = __expf(xa[j] - m);
  red[j] = e;
  __syncthreads();
  for (int off = 32; off > 0; off >>= 1){
    if (j < off) red[j] += red[j+off];
    __syncthreads();
  }
  att[(size_t)row*64 + j] = e / red[0];
}

// ------------- Wcomb[g][c*27+s] = sum_{o,p valid} w_ih[g,(o,p)] * conv_w[o,c,s-p+1] -------------
__global__ __launch_bounds__(256) void wcomb_kernel(const float* __restrict__ conv_w,
    const float* __restrict__ w_ih, f16* __restrict__ wc)
{
  const int g  = threadIdx.x;
  const int cs = blockIdx.x;
  const int c = cs / 27, s = cs % 27;
  const int sz = s/9, sy = (s%9)/3, sx = s%3;
  float acc = 0.f;
  for (int pz = imax(0, sz-1); pz <= imin(2, sz+1); ++pz){
    const int dz = sz - pz + 1;
    for (int py = imax(0, sy-1); py <= imin(2, sy+1); ++py){
      const int dy = sy - py + 1;
      for (int px = imax(0, sx-1); px <= imin(2, sx+1); ++px){
        const int dx = sx - px + 1;
        const int pp = (pz*3 + py)*3 + px;
        const int qq = (dz*3 + dy)*3 + dx;
        const float* wih = w_ih + (size_t)g*KV + pp;
        const float* cw  = conv_w + c*27 + qq;
        #pragma unroll 8
        for (int o = 0; o < 64; ++o)
          acc += wih[o*27] * cw[o*KV];
      }
    }
  }
  wc[(size_t)g*KV + cs] = (f16)acc;
}

// bias_comb[g] = b_ih + b_hh + sum_{o,p} w_ih[g,(o,p)] * conv_b[o]
__global__ __launch_bounds__(256) void bias_kernel(const float* __restrict__ w_ih,
    const float* __restrict__ conv_b, const float* __restrict__ b_ih,
    const float* __restrict__ b_hh, float* __restrict__ biasc)
{
  const int g = threadIdx.x;
  float acc = b_ih[g] + b_hh[g];
  for (int o = 0; o < 64; ++o){
    const float cb = conv_b[o];
    const float* w = w_ih + (size_t)g*KV + o*27;
    #pragma unroll
    for (int p = 0; p < 27; ++p) acc += w[p] * cb;
  }
  biasc[g] = acc;
}

// ---------- main WMMA GEMM: gpre[16384,256] = f16(x*att)[16384,1728] x Wcomb^T + bias ----------
__global__ __launch_bounds__(256) void gemm_kernel(const float* __restrict__ x,
    const float* __restrict__ att, const f16* __restrict__ wc,
    const float* __restrict__ biasc, float* __restrict__ gpre)
{
  __shared__ __align__(16) f16 Atile[64*64];   // [M=64][K=64]
  __shared__ __align__(16) f16 Btile[256*64];  // [N=256][K=64]
  const int tid  = threadIdx.x;
  const int lane = tid & 31;
  const int w    = tid >> 5;   // 8 waves
  const int wm   = w & 3;      // 4 M sub-tiles of 16
  const int wn   = w >> 2;     // 2 N halves of 128
  const int m0   = blockIdx.x * 64;

  v8f acc[8] = {};

  const int arow = tid >> 2;        // 0..63
  const int aq   = (tid & 3) * 16;  // K-chunk within tile

  // LDS byte offset of this thread's B-tile row (low 32 bits of generic ptr)
  const unsigned ldsB = (unsigned)(uintptr_t)(Btile + tid*64);

  for (int k0 = 0; k0 < KV; k0 += 64){
    // B tile: async DMA global f16 -> LDS (no VGPR round trip), 8 x 16B per thread
    {
      const f16* gsrc = wc + (size_t)tid*KV + k0;
      #pragma unroll
      for (int j = 0; j < 8; ++j){
        asm volatile("global_load_async_to_lds_b128 %0, %1, off"
                     :: "v"(ldsB + j*16), "v"(gsrc + j*8) : "memory");
      }
    }
    // A tile: read x f32, scale by att, convert to f16 (overlaps with async B DMA)
    {
      const float* src = x + (size_t)(m0 + arow)*KV + k0 + aq;
      __builtin_prefetch(src + 64, 0, 0);      // global_prefetch for next K-block
      f16* dst = Atile + arow*64 + aq;
      const float* arow_att = att + (size_t)(m0 + arow)*64;
      #pragma unroll
      for (int j = 0; j < 16; ++j){
        const int kg = k0 + aq + j;
        dst[j] = (f16)(src[j] * arow_att[kg / 27]);
      }
    }
    asm volatile("s_wait_asynccnt 0x0" ::: "memory");  // async B copy done
    __syncthreads();
    #pragma unroll
    for (int kk = 0; kk < 64; kk += 32){
      const v16h a = frag_a16(Atile + (wm*16 + (lane & 15))*64 + kk + ((lane >> 4)*8));
      #pragma unroll
      for (int j = 0; j < 8; ++j){
        const int n = wn*128 + j*16 + (lane & 15);
        const v16h b = frag_a16(Btile + n*64 + kk + ((lane >> 4)*8));
        acc[j] = __builtin_amdgcn_wmma_f32_16x16x32_f16(
            false, a, false, b, (short)0, acc[j], false, false);
      }
    }
    __syncthreads();
  }
  #pragma unroll
  for (int j = 0; j < 8; ++j){
    const int col = wn*128 + j*16 + (lane & 15);
    const float bsum = biasc[col];
    #pragma unroll
    for (int v = 0; v < 8; ++v){
      const int r = m0 + wm*16 + v + ((lane >> 4)*8);
      gpre[(size_t)r*NG + col] = acc[j][v] + bsum;
    }
  }
}

// ---------- LSTM recurrence: 8 blocks x 16 batch rows, w_hh resident as WMMA B-frags ----------
__global__ __launch_bounds__(512) void lstm_kernel(const float* __restrict__ gpre,
    const float* __restrict__ w_hh, const float* __restrict__ fc_w,
    const float* __restrict__ fc_b, float* __restrict__ out)
{
  __shared__ __align__(16) f16 hbuf[16*64];   // h as f16 [16 rows][64]
  __shared__ float gbuf[16*256];              // raw gates
  const int tid  = threadIdx.x;
  const int lane = tid & 31;
  const int wv   = tid >> 5;                  // 16 waves = 16 N-tiles
  const int b0   = blockIdx.x * 16;
  const int ncol = wv*16 + (lane & 15);       // global gate column

  // preload w_hh as two B-fragments per wave (constant over T)
  v16h bf[2];
  #pragma unroll
  for (int h = 0; h < 2; ++h){
    v16h t;
    #pragma unroll
    for (int e = 0; e < 16; ++e){
      const int kk = h*32 + ((lane >> 4)*8) + (e & 7) + ((e >> 3)*16);
      t[e] = (f16)w_hh[(size_t)ncol*64 + kk];
    }
    bf[h] = t;
  }
  for (int i = tid; i < 16*64; i += 512) hbuf[i] = (f16)0.f;
  float cst[2]  = {0.f, 0.f};
  float hsum[2] = {0.f, 0.f};
  __syncthreads();

  for (int t = 0; t < Tsz; ++t){
    // C operand seeded from precomputed gates
    v8f acc;
    #pragma unroll
    for (int v = 0; v < 8; ++v){
      const int rl = v + ((lane >> 4)*8);
      acc[v] = gpre[(size_t)((b0 + rl)*Tsz + t)*NG + ncol];
    }
    const v16h a0 = frag_a16(hbuf + (lane & 15)*64 + 0  + ((lane >> 4)*8));
    const v16h a1 = frag_a16(hbuf + (lane & 15)*64 + 32 + ((lane >> 4)*8));
    acc = __builtin_amdgcn_wmma_f32_16x16x32_f16(false, a0, false, bf[0], (short)0, acc, false, false);
    acc = __builtin_amdgcn_wmma_f32_16x16x32_f16(false, a1, false, bf[1], (short)0, acc, false, false);
    #pragma unroll
    for (int v = 0; v < 8; ++v){
      const int rl = v + ((lane >> 4)*8);
      gbuf[rl*256 + ncol] = acc[v];
    }
    __syncthreads();
    // elementwise cell update: thread owns (r,hc) pairs idx = tid + e*512
    #pragma unroll
    for (int e = 0; e < 2; ++e){
      const int idx = tid + e*512;
      const int r = idx >> 6, hc = idx & 63;
      const float gi = gbuf[r*256 + hc];
      const float gf = gbuf[r*256 +  64 + hc];
      const float gg = gbuf[r*256 + 128 + hc];
      const float go = gbuf[r*256 + 192 + hc];
      const float cn = sigf(gf)*cst[e] + sigf(gi)*tanhf(gg);
      const float hn = sigf(go)*tanhf(cn);
      cst[e] = cn;
      hsum[e] += hn;
      hbuf[r*64 + hc] = (f16)hn;
    }
    __syncthreads();
  }
  // mean over T then final fc (HID -> 3)
  #pragma unroll
  for (int e = 0; e < 2; ++e){
    const int idx = tid + e*512;
    gbuf[idx] = hsum[e] * (1.f/(float)Tsz);   // reuse as hmean[16][64]
  }
  __syncthreads();
  if (tid < 48){
    const int r = tid / 3, j = tid % 3;
    float s = fc_b[j];
    #pragma unroll 8
    for (int hc = 0; hc < 64; ++hc) s += gbuf[r*64 + hc] * fc_w[j*64 + hc];
    out[(b0 + r)*3 + j] = s;
  }
}

extern "C" void kernel_launch(void* const* d_in, const int* in_sizes, int n_in,
                              void* d_out, int out_size, void* d_ws, size_t ws_size,
                              hipStream_t stream)
{
  (void)in_sizes; (void)n_in; (void)out_size; (void)ws_size;
  const float* x     = (const float*)d_in[0];
  const float* fc1w  = (const float*)d_in[1];
  const float* fc1b  = (const float*)d_in[2];
  const float* fc2w  = (const float*)d_in[3];
  const float* fc2b  = (const float*)d_in[4];
  const float* convw = (const float*)d_in[5];
  const float* convb = (const float*)d_in[6];
  const float* wih   = (const float*)d_in[7];
  const float* whh   = (const float*)d_in[8];
  const float* bih   = (const float*)d_in[9];
  const float* bhh   = (const float*)d_in[10];
  const float* fcw   = (const float*)d_in[11];
  const float* fcb   = (const float*)d_in[12];
  float* out = (float*)d_out;

  char* ws = (char*)d_ws;
  float* att   = (float*)ws;                              // 16384*64*4  = 4,194,304 B
  f16*   wc    = (f16*)(ws + 4194304);                    // 256*1728*2  =   884,736 B
  float* biasc = (float*)(ws + 4194304 + 884736);         // 256*4       =     1,024 B
  float* gpre  = (float*)(ws + 4194304 + 884736 + 1024);  // 16384*256*4 = 16,777,216 B

  attn_kernel <<<MT,     64, 0, stream>>>(x, fc1w, fc1b, fc2w, fc2b, att);
  wcomb_kernel<<<KV,    256, 0, stream>>>(convw, wih, wc);
  bias_kernel <<<1,     256, 0, stream>>>(wih, convb, bih, bhh, biasc);
  gemm_kernel <<<MT/64, 256, 0, stream>>>(x, att, wc, biasc, gpre);
  lstm_kernel <<<Bsz/16, 512, 0, stream>>>(gpre, whh, fcw, fcb, out);
}
